// CausalSelfAttention_9328668967077
// MI455X (gfx1250) — matrix operand order repaired and checked by reading, not compiled, not run on hardware
//
#include <hip/hip_runtime.h>
#include <hip/hip_bf16.h>
#include <math.h>

// ---------------------------------------------------------------------------
// CDNA5 (gfx1250) causal self-attention:  qkv GEMM -> flash attention -> proj
// v_wmma_f32_16x16x32_bf16 matmuls, async global->LDS staging (ASYNCcnt),
// TDM tensor_load_to_lds for the strided V tile (TENSORcnt).
// ---------------------------------------------------------------------------

typedef __attribute__((ext_vector_type(16))) __bf16        v16bf;
typedef __attribute__((ext_vector_type(8)))  float         v8f;
typedef __attribute__((ext_vector_type(4)))  unsigned int  v4u;
typedef __attribute__((ext_vector_type(8)))  int           v8i;
typedef __attribute__((ext_vector_type(4)))  int           v4i;

#define B_  2
#define T_  2048
#define C_  1024
#define NH_ 16
#define HD_ 64
#define M1_ (B_ * T_)      // 4096 rows of x
#define N3_ (3 * C_)       // 3072

union FragU { v16bf v; unsigned int u[8]; };

static __device__ __forceinline__ unsigned short f2bf(float x) {
  unsigned u = __float_as_uint(x);
  unsigned r = u + 0x7FFFu + ((u >> 16) & 1u);   // round to nearest even
  return (unsigned short)(r >> 16);
}

static __device__ __forceinline__ v8f wmma_bf16(v16bf a, v16bf b, v8f c) {
  return __builtin_amdgcn_wmma_f32_16x16x32_bf16(false, a, false, b,
                                                 (short)0, c, false, false);
}

// Async global->LDS 16B copy (per-lane). Tracked by ASYNCcnt.
static __device__ __forceinline__ void async_b128(void* lds, const void* g) {
  unsigned ldso = (unsigned)(unsigned long long)lds;
  asm volatile("global_load_async_to_lds_b128 %0, %1, off"
               :: "v"(ldso), "v"(g) : "memory");
}
static __device__ __forceinline__ void wait_async0() {
  asm volatile("s_wait_asynccnt 0x0" ::: "memory");
}

// TDM: DMA a 2D tile (tileX x tileY elements of 2 bytes, source row stride
// strideElems) from global memory into contiguous LDS.  One instruction per
// wave; tracked by TENSORcnt.  D# layout per CDNA5 ISA.
// This toolchain exposes the 6-arg builtin:
//   (v4u g0, v8i g1, v4i g2, v4i g3, v8i extra, i32 cpol)
static __device__ __forceinline__ void tdm_load_2d_b16(
    void* lds, const void* g, unsigned tileX, unsigned tileY,
    unsigned dim0, unsigned dim1, unsigned strideElems) {
  unsigned long long ga = (unsigned long long)g;
  unsigned ldso = (unsigned)(unsigned long long)lds;
  v4u g0;
  g0.x = 1u;                                    // count=1, user mode
  g0.y = ldso;                                  // lds_addr
  g0.z = (unsigned)ga;                          // global_addr[31:0]
  g0.w = (unsigned)((ga >> 32) & 0x1FFFFFFu)    // global_addr[56:32]
         | (2u << 30);                          // type=2 ("image")
  v8i g1;
  g1[0] = (int)(1u << 16);                      // data_size=1 (2 bytes)
  g1[1] = (int)(dim0 << 16);                    // tensor_dim0[15:0] @ bits 63:48
  g1[2] = (int)((dim0 >> 16) | (dim1 << 16));   // dim0[31:16] | dim1[15:0]
  g1[3] = (int)((dim1 >> 16) | (tileX << 16));  // dim1[31:16] | tile_dim0
  g1[4] = (int)tileY;                           // tile_dim1 (tile_dim2=0)
  g1[5] = (int)strideElems;                     // tensor_dim0_stride[31:0]
  g1[6] = 0;                                    // stride[47:32], dim1_stride lo
  g1[7] = 0;
  v4i g2;
  g2[0] = 1; g2[1] = 0; g2[2] = 0; g2[3] = 0;   // tensor_dim2=1, rest unused
  v4i g3;
  g3[0] = 0; g3[1] = 0; g3[2] = 0; g3[3] = 0;
  v8i gx;
  gx[0] = 0; gx[1] = 0; gx[2] = 0; gx[3] = 0;
  gx[4] = 0; gx[5] = 0; gx[6] = 0; gx[7] = 0;
  __builtin_amdgcn_tensor_load_to_lds(g0, g1, g2, g3, gx, 0);
}
static __device__ __forceinline__ void wait_tensor0() {
  __builtin_amdgcn_s_wait_tensorcnt(0);
}

// A-fragment: 16x32 bf16.  lane16 = M row; element j pair at
//   k = 2j + (j>=4 ? 8 : 0) + 8*laneHalf   (ISA 7.12.2, 16-bit A 16x32)
static __device__ __forceinline__ v16bf load_frag_a(const unsigned short* base,
                                                    int ld, int lane16, int half) {
  FragU f;
#pragma unroll
  for (int j = 0; j < 8; ++j) {
    int k = 2 * j + ((j >= 4) ? 8 : 0) + 8 * half;
    f.u[j] = *(const unsigned int*)(base + lane16 * ld + k);
  }
  return f.v;
}

// B-fragment: 32x16 bf16, per-lane column N = lane16, stored N-major.
// same K order as A:  k = 2j + (j>=4 ? 8 : 0) + 8*laneHalf
static __device__ __forceinline__ v16bf load_frag_b(const unsigned short* base,
                                                    int ld, int lane16, int half) {
  FragU f;
#pragma unroll
  for (int j = 0; j < 8; ++j) {
    int k = 2 * j + ((j >= 4) ? 8 : 0) + 8 * half;
    f.u[j] = *(const unsigned int*)(base + lane16 * ld + k);
  }
  return f.v;
}

// ---------------------------------------------------------------------------
// Kernel 1: qkv = x @ Wqkv  (f32 in, bf16 out, scattered to q/k/v layouts)
// Block tile 128(M) x 64(N), K-step 32, 8 waves (each 32x32 output).
// ---------------------------------------------------------------------------
__global__ __launch_bounds__(256) void qkv_gemm_kernel(
    const float* __restrict__ x, const float* __restrict__ W,
    unsigned short* __restrict__ q_ws, unsigned short* __restrict__ k_ws,
    unsigned short* __restrict__ v_ws) {
  __shared__ unsigned short As[128 * 32];  // [m][k]
  __shared__ unsigned short Bs[64 * 32];   // transposed: [n][k]

  const int tid = threadIdx.x;
  const int wave = tid >> 5, lane = tid & 31;
  const int lane16 = lane & 15, half = lane >> 4;
  const int wm = wave & 3, wn = wave >> 2;
  const int rowBase = blockIdx.y * 128;
  const int colBase = blockIdx.x * 64;

  v8f acc[2][2];
#pragma unroll
  for (int i = 0; i < 2; ++i)
#pragma unroll
    for (int j = 0; j < 2; ++j)
      acc[i][j] = v8f{0.f, 0.f, 0.f, 0.f, 0.f, 0.f, 0.f, 0.f};

  for (int k0 = 0; k0 < C_; k0 += 32) {
    // prefetch next K-step tiles into GL2 while we convert this one
    if (k0 + 32 < C_) {
      __builtin_prefetch(x + (size_t)(rowBase + (tid >> 1)) * C_ + k0 + 32, 0, 1);
      __builtin_prefetch(W + (size_t)(k0 + 32 + (tid & 31)) * N3_ + colBase, 0, 1);
    }
    // stage A: 128x32 f32 -> bf16  (1024 float4, 4 per thread)
#pragma unroll
    for (int i = 0; i < 4; ++i) {
      int idx = tid + i * 256;
      int row = idx >> 3, c4 = (idx & 7) * 4;
      float4 f = *(const float4*)(x + (size_t)(rowBase + row) * C_ + k0 + c4);
      unsigned lo = (unsigned)f2bf(f.x) | ((unsigned)f2bf(f.y) << 16);
      unsigned hi = (unsigned)f2bf(f.z) | ((unsigned)f2bf(f.w) << 16);
      unsigned* dst = (unsigned*)&As[row * 32 + c4];
      dst[0] = lo; dst[1] = hi;
    }
    // stage B transposed: 32(k) x 64(n) f32 -> Bs[n][k]
#pragma unroll
    for (int i = 0; i < 8; ++i) {
      int idx = tid + i * 256;
      int kk = idx >> 6, n = idx & 63;
      Bs[n * 32 + kk] = f2bf(W[(size_t)(k0 + kk) * N3_ + colBase + n]);
    }
    __syncthreads();

    v16bf a0 = load_frag_a(As + (wm * 32 + 0) * 32, 32, lane16, half);
    v16bf a1 = load_frag_a(As + (wm * 32 + 16) * 32, 32, lane16, half);
    v16bf b0 = load_frag_b(Bs + (wn * 32 + 0) * 32, 32, lane16, half);
    v16bf b1 = load_frag_b(Bs + (wn * 32 + 16) * 32, 32, lane16, half);
    acc[0][0] = wmma_bf16(a0, b0, acc[0][0]);
    acc[0][1] = wmma_bf16(a0, b1, acc[0][1]);
    acc[1][0] = wmma_bf16(a1, b0, acc[1][0]);
    acc[1][1] = wmma_bf16(a1, b1, acc[1][1]);
    __syncthreads();
  }

  // epilogue: scatter into q / k / v
#pragma unroll
  for (int mm = 0; mm < 2; ++mm)
#pragma unroll
    for (int nn = 0; nn < 2; ++nn)
#pragma unroll
      for (int r = 0; r < 8; ++r) {
        int grow = rowBase + wm * 32 + mm * 16 + r + 8 * half;
        int gcol = colBase + wn * 32 + nn * 16 + lane16;
        float val = acc[mm][nn][r];
        int sec = gcol >> 10;          // 0=q 1=k 2=v (uniform per block)
        int cc = gcol & (C_ - 1);
        int h = cc >> 6, d = cc & 63;
        int b = grow >> 11, t = grow & (T_ - 1);
        if (sec == 0) {
          q_ws[((size_t)(b * NH_ + h) * T_ + t) * HD_ + d] = f2bf(val * 0.125f);
        } else if (sec == 1) {
          k_ws[((size_t)(b * NH_ + h) * T_ + t) * HD_ + d] = f2bf(val);
        } else {
          v_ws[((size_t)(b * NH_ + h) * HD_ + d) * T_ + t] = f2bf(val);
        }
      }
}

// ---------------------------------------------------------------------------
// Kernel 2: flash attention per (b, h, 64-row q tile).  Online softmax over
// 64-key tiles.  Q/K tiles (contiguous 8KB) staged via async global->LDS;
// V tile (strided 2D: 64 rows x 128B, row stride 4KB) staged via one TDM
// tensor_load_to_lds issued by wave 0.
// ---------------------------------------------------------------------------
#define SFLD 68   // padded f32 stride: kills the lane16-31 bank aliasing

__global__ __launch_bounds__(256) void flash_attn_kernel(
    const unsigned short* __restrict__ q_ws,
    const unsigned short* __restrict__ k_ws,
    const unsigned short* __restrict__ v_ws,
    unsigned short* __restrict__ y_ws) {
  __shared__ unsigned short Qs[64 * 64];   // bf16 Q tile  [q][d]
  __shared__ unsigned short Ks[64 * 64];   // bf16 K tile  [key][d]
  __shared__ unsigned short Vs[64 * 64];   // bf16 Vt tile [d][key]
  __shared__ float          Sf[64 * SFLD]; // f32 scores (padded)
  __shared__ unsigned short Ps[64 * 64];   // bf16 probabilities
  __shared__ float m_s[64], l_s[64], sc_s[64];

  const int tid = threadIdx.x;
  const int wave = tid >> 5, lane = tid & 31;
  const int lane16 = lane & 15, half = lane >> 4;
  const int qt = blockIdx.x, h = blockIdx.y, b = blockIdx.z;
  const int q0 = qt * 64;

  const unsigned short* qh  = q_ws + (size_t)(b * NH_ + h) * T_ * HD_;
  const unsigned short* kh  = k_ws + (size_t)(b * NH_ + h) * T_ * HD_;
  const unsigned short* vth = v_ws + (size_t)(b * NH_ + h) * HD_ * T_;  // [d][t]

  // async-load Q tile (64x64 bf16 = 512 x 16B chunks, 2 per thread)
#pragma unroll
  for (int i = 0; i < 2; ++i) {
    int idx = tid + i * 256;
    int row = idx >> 3, c8 = (idx & 7) * 8;
    async_b128(Qs + row * 64 + c8, qh + (size_t)(q0 + row) * HD_ + c8);
  }
  if (tid < 64) { m_s[tid] = -__builtin_inff(); l_s[tid] = 0.f; }
  wait_async0();
  __syncthreads();

  const int sm = wave & 3;            // 16-row block of this wave
  const int snb = (wave >> 2) * 2;    // first of two 16-col blocks

  v8f yacc[2];
  yacc[0] = v8f{0.f, 0.f, 0.f, 0.f, 0.f, 0.f, 0.f, 0.f};
  yacc[1] = v8f{0.f, 0.f, 0.f, 0.f, 0.f, 0.f, 0.f, 0.f};

  for (int kt = 0; kt <= qt; ++kt) {
    const int k0 = kt * 64;
    // ---- stage K tile (async, contiguous) + V tile (TDM, strided 2D) ----
    if (wave == 0) {
      // tile: 64 keys (X) x 64 d-rows (Y), source row stride T_ elements
      tdm_load_2d_b16(Vs, vth + k0, 64, 64, (unsigned)T_, (unsigned)HD_,
                      (unsigned)T_);
    }
#pragma unroll
    for (int i = 0; i < 2; ++i) {
      int idx = tid + i * 256;
      int row = idx >> 3, c8 = (idx & 7) * 8;
      async_b128(Ks + row * 64 + c8, kh + (size_t)(k0 + row) * HD_ + c8);
    }
    if (kt + 1 <= qt)   // prefetch next key tile toward GL2
      __builtin_prefetch(kh + (size_t)(k0 + 64 + (tid >> 2)) * HD_, 0, 1);
    wait_async0();
    if (wave == 0) wait_tensor0();
    __syncthreads();

    // ---- phase 1: S = Q @ K^T (each wave: 2 tiles of 16x16) ----
#pragma unroll
    for (int nn = 0; nn < 2; ++nn) {
      v8f s = v8f{0.f, 0.f, 0.f, 0.f, 0.f, 0.f, 0.f, 0.f};
#pragma unroll
      for (int kk = 0; kk < HD_; kk += 32) {
        v16bf a  = load_frag_a(Qs + (sm * 16) * 64 + kk, 64, lane16, half);
        v16bf bb = load_frag_b(Ks + ((snb + nn) * 16) * 64 + kk, 64, lane16, half);
        s = wmma_bf16(a, bb, s);
      }
#pragma unroll
      for (int r = 0; r < 8; ++r) {
        int rr = sm * 16 + r + 8 * half;
        int cc = (snb + nn) * 16 + lane16;
        float v = s[r];
        if (kt == qt && cc > rr) v = -__builtin_inff();  // causal mask (diag)
        Sf[rr * SFLD + cc] = v;
      }
    }
    __syncthreads();

    // ---- phase 2: online softmax (4 threads per row) ----
    {
      const int row = tid >> 2, part = tid & 3, cb = part * 16;
      float mt = -__builtin_inff();
#pragma unroll
      for (int c = 0; c < 16; ++c) mt = fmaxf(mt, Sf[row * SFLD + cb + c]);
      mt = fmaxf(mt, __shfl_xor(mt, 1, 32));
      mt = fmaxf(mt, __shfl_xor(mt, 2, 32));
      float m_old = m_s[row];
      float m_new = fmaxf(m_old, mt);
      float scl = __expf(m_old - m_new);
      float psum = 0.f;
#pragma unroll
      for (int c = 0; c < 16; ++c) {
        float e = __expf(Sf[row * SFLD + cb + c] - m_new);
        Ps[row * 64 + cb + c] = f2bf(e);
        psum += e;
      }
      psum += __shfl_xor(psum, 1, 32);
      psum += __shfl_xor(psum, 2, 32);
      if (part == 0) {
        m_s[row] = m_new;
        l_s[row] = l_s[row] * scl + psum;
        sc_s[row] = scl;
      }
    }
    __syncthreads();

    // ---- phase 3: rescale Y, then Y += P @ V ----
    {
      float srow[8];
#pragma unroll
      for (int r = 0; r < 8; ++r) srow[r] = sc_s[sm * 16 + r + 8 * half];
#pragma unroll
      for (int nn = 0; nn < 2; ++nn)
#pragma unroll
        for (int r = 0; r < 8; ++r) yacc[nn][r] *= srow[r];
#pragma unroll
      for (int nn = 0; nn < 2; ++nn) {
#pragma unroll
        for (int kk = 0; kk < 64; kk += 32) {
          v16bf a  = load_frag_a(Ps + (sm * 16) * 64 + kk, 64, lane16, half);
          v16bf bb = load_frag_b(Vs + ((snb + nn) * 16) * 64 + kk, 64, lane16, half);
          yacc[nn] = wmma_bf16(a, bb, yacc[nn]);
        }
      }
    }
    __syncthreads();
  }

  // finalize: y = Y / l, write bf16 [B*T, C] with head interleave undone
#pragma unroll
  for (int nn = 0; nn < 2; ++nn)
#pragma unroll
    for (int r = 0; r < 8; ++r) {
      int rr = sm * 16 + r + 8 * half;
      float yv = yacc[nn][r] * (1.f / l_s[rr]);
      int d = (snb + nn) * 16 + lane16;
      y_ws[((size_t)(b * T_ + q0 + rr)) * C_ + h * HD_ + d] = f2bf(yv);
    }
}

// ---------------------------------------------------------------------------
// Kernel 3: out = y @ Wproj  (bf16 A, f32 W -> f32 out).  A tile is a pure
// bf16 copy -> async global->LDS; B tile converted/transposed on VALU.
// ---------------------------------------------------------------------------
__global__ __launch_bounds__(256) void proj_gemm_kernel(
    const unsigned short* __restrict__ y_ws, const float* __restrict__ W,
    float* __restrict__ out) {
  __shared__ unsigned short As[128 * 32];
  __shared__ unsigned short Bs[64 * 32];

  const int tid = threadIdx.x;
  const int wave = tid >> 5, lane = tid & 31;
  const int lane16 = lane & 15, half = lane >> 4;
  const int wm = wave & 3, wn = wave >> 2;
  const int rowBase = blockIdx.y * 128;
  const int colBase = blockIdx.x * 64;

  v8f acc[2][2];
#pragma unroll
  for (int i = 0; i < 2; ++i)
#pragma unroll
    for (int j = 0; j < 2; ++j)
      acc[i][j] = v8f{0.f, 0.f, 0.f, 0.f, 0.f, 0.f, 0.f, 0.f};

  for (int k0 = 0; k0 < C_; k0 += 32) {
    // stage A: async copy 128x32 bf16 (512 x 16B chunks, 2 per thread)
#pragma unroll
    for (int i = 0; i < 2; ++i) {
      int idx = tid + i * 256;
      int row = idx >> 2, c8 = (idx & 3) * 8;
      async_b128(As + row * 32 + c8,
                 y_ws + (size_t)(rowBase + row) * C_ + k0 + c8);
    }
    // stage B transposed: 32(k) x 64(n) f32 -> Bs[n][k]
#pragma unroll
    for (int i = 0; i < 8; ++i) {
      int idx = tid + i * 256;
      int kk = idx >> 6, n = idx & 63;
      Bs[n * 32 + kk] = f2bf(W[(size_t)(k0 + kk) * C_ + colBase + n]);
    }
    if (k0 + 32 < C_)
      __builtin_prefetch(W + (size_t)(k0 + 32 + (tid & 31)) * C_ + colBase, 0, 1);
    wait_async0();
    __syncthreads();

    v16bf a0 = load_frag_a(As + (wm * 32 + 0) * 32, 32, lane16, half);
    v16bf a1 = load_frag_a(As + (wm * 32 + 16) * 32, 32, lane16, half);
    v16bf b0 = load_frag_b(Bs + (wn * 32 + 0) * 32, 32, lane16, half);
    v16bf b1 = load_frag_b(Bs + (wn * 32 + 16) * 32, 32, lane16, half);
    acc[0][0] = wmma_bf16(a0, b0, acc[0][0]);
    acc[0][1] = wmma_bf16(a0, b1, acc[0][1]);
    acc[1][0] = wmma_bf16(a1, b0, acc[1][0]);
    acc[1][1] = wmma_bf16(a1, b1, acc[1][1]);
    __syncthreads();
  }

#pragma unroll
  for (int mm = 0; mm < 2; ++mm)
#pragma unroll
    for (int nn = 0; nn < 2; ++nn)
#pragma unroll
      for (int r = 0; r < 8; ++r) {
        int grow = rowBase + wm * 32 + mm * 16 + r + 8 * half;
        int gcol = colBase + wn * 32 + nn * 16 + lane16;
        out[(size_t)grow * C_ + gcol] = acc[mm][nn][r];
      }
}

// ---------------------------------------------------------------------------
extern "C" void kernel_launch(void* const* d_in, const int* in_sizes, int n_in,
                              void* d_out, int out_size, void* d_ws, size_t ws_size,
                              hipStream_t stream) {
  const float* x     = (const float*)d_in[0];  // [B,T,C]
  const float* Wqkv  = (const float*)d_in[1];  // [C,3C]
  const float* Wproj = (const float*)d_in[2];  // [C,C]
  float* out = (float*)d_out;                  // [B,T,C]

  // workspace layout (bf16): q | k | v(transposed) | y   (8 MB each = 32 MB)
  unsigned short* q_ws = (unsigned short*)d_ws;
  unsigned short* k_ws = q_ws + (size_t)B_ * NH_ * T_ * HD_;
  unsigned short* v_ws = k_ws + (size_t)B_ * NH_ * T_ * HD_;
  unsigned short* y_ws = v_ws + (size_t)B_ * NH_ * T_ * HD_;

  dim3 blk(256);
  dim3 g1(N3_ / 64, M1_ / 128);           // 48 x 32
  qkv_gemm_kernel<<<g1, blk, 0, stream>>>(x, Wqkv, q_ws, k_ws, v_ws);

  dim3 g2(T_ / 64, NH_, B_);              // 32 x 16 x 2
  flash_attn_kernel<<<g2, blk, 0, stream>>>(q_ws, k_ws, v_ws, y_ws);

  dim3 g3(C_ / 64, M1_ / 128);            // 16 x 32
  proj_gemm_kernel<<<g3, blk, 0, stream>>>(y_ws, Wproj, out);
}
